// MultiHeadAttentionWithRoPE_27582279975572
// MI455X (gfx1250) — matrix operand order repaired and checked by reading, not compiled, not run on hardware
//
#include <hip/hip_runtime.h>

// ---------------------------------------------------------------------------
// MultiHeadAttentionWithRoPE for MI455X (gfx1250, wave32, WMMA 16x16x32 f16)
// B=2, S=2048, D=1024, H=16, HD=64
// ---------------------------------------------------------------------------

typedef _Float16 f16;
typedef __attribute__((ext_vector_type(16))) _Float16 v16h;
typedef __attribute__((ext_vector_type(8)))  _Float16 v8h;
typedef __attribute__((ext_vector_type(4)))  _Float16 v4h;
typedef __attribute__((ext_vector_type(8)))  float    v8f;
typedef __attribute__((ext_vector_type(4)))  float    v4f;

#define CB  2
#define CS  2048
#define CD  1024
#define CH  16
#define CHD 64

__device__ __forceinline__ v16h cat8(v8h lo, v8h hi) {
  return __builtin_shufflevector(lo, hi, 0,1,2,3,4,5,6,7,8,9,10,11,12,13,14,15);
}

__device__ __forceinline__ v8f wmma_f16(v16h a, v16h b, v8f c) {
  // D = A(16x32 f16) * B(32x16 f16) + C(16x16 f32)
  return __builtin_amdgcn_wmma_f32_16x16x32_f16(false, a, false, b, (short)0, c,
                                                false, false);
}

// Async DMA: global (16B per lane) -> LDS, tracked by ASYNCcnt.
__device__ __forceinline__ void async_b128(const f16* g, f16* l) {
  unsigned lds = (unsigned)(unsigned long long)l;  // low 32 bits = LDS offset
  asm volatile("global_load_async_to_lds_b128 %0, %1, off"
               :: "v"(lds), "v"((unsigned long long)g) : "memory");
}

// ---------------------------------------------------------------------------
// f32 -> f16 convert (vectorized x4)
// ---------------------------------------------------------------------------
__global__ __launch_bounds__(256) void cvt_f32_f16(const float* __restrict__ src,
                                                   f16* __restrict__ dst, int n4) {
  int i = blockIdx.x * blockDim.x + threadIdx.x;
  if (i >= n4) return;
  v4f x = *(const v4f*)(src + 4 * (size_t)i);
  v4h y;
  y.x = (f16)x.x; y.y = (f16)x.y; y.z = (f16)x.z; y.w = (f16)x.w;
  *(v4h*)(dst + 4 * (size_t)i) = y;
}

// ---------------------------------------------------------------------------
// WMMA GEMM: out[M=4096, N=1024] = A[4096,1024] * W[1024,1024]^T
// One wave computes a 32x64 block (2 m-tiles x 4 n-tiles): A-frags reused 4x,
// B-frags reused 2x, 8 WMMAs per 32-wide k-step.
// MODE 0: f16 output, natural [row, col] layout       (Q, K)
// MODE 1: f16 output, transposed V layout [B,H,HD,S]  (V)
// MODE 2: f32 output + bias                           (final projection)
// ---------------------------------------------------------------------------
template <int MODE>
__global__ __launch_bounds__(256) void gemm_wmma(const f16* __restrict__ A,
                                                 const f16* __restrict__ W,
                                                 f16* __restrict__ oh,
                                                 float* __restrict__ of,
                                                 const float* __restrict__ bias) {
  const int NG = CD / 64;  // 16 groups of 4 n-tiles
  int wid  = blockIdx.x * (blockDim.x >> 5) + (threadIdx.x >> 5);
  int lane = threadIdx.x & 31;
  int mg = wid / NG, ng = wid % NG;
  int m0 = mg * 32, n0 = ng * 64;
  int lm = lane & 15, sel = lane >> 4;

  v8f acc[2][4];
#pragma unroll
  for (int i = 0; i < 2; i++)
#pragma unroll
    for (int j = 0; j < 4; j++) acc[i][j] = (v8f){0.f,0.f,0.f,0.f,0.f,0.f,0.f,0.f};

  // A layout: lane holds row M = lane&15; K-halves selected by lane>>4
  const f16* arow0 = A + (size_t)(m0 + lm) * CD;
  const f16* arow1 = A + (size_t)(m0 + 16 + lm) * CD;
  // B = W^T: lane holds column n (== W row), K contiguous
  const f16* wrow0 = W + (size_t)(n0 + 0 * 16 + lm) * CD + sel * 16;
  const f16* wrow1 = W + (size_t)(n0 + 1 * 16 + lm) * CD + sel * 16;
  const f16* wrow2 = W + (size_t)(n0 + 2 * 16 + lm) * CD + sel * 16;
  const f16* wrow3 = W + (size_t)(n0 + 3 * 16 + lm) * CD + sel * 16;

#pragma unroll 2
  for (int k = 0; k < CD; k += 32) {
    v16h a0 = *(const v16h*)(arow0 + k + sel * 16);
    v16h a1 = *(const v16h*)(arow1 + k + sel * 16);
    v16h b0 = *(const v16h*)(wrow0 + k);
    v16h b1 = *(const v16h*)(wrow1 + k);
    v16h b2 = *(const v16h*)(wrow2 + k);
    v16h b3 = *(const v16h*)(wrow3 + k);
    acc[0][0] = wmma_f16(a0, b0, acc[0][0]);
    acc[0][1] = wmma_f16(a0, b1, acc[0][1]);
    acc[0][2] = wmma_f16(a0, b2, acc[0][2]);
    acc[0][3] = wmma_f16(a0, b3, acc[0][3]);
    acc[1][0] = wmma_f16(a1, b0, acc[1][0]);
    acc[1][1] = wmma_f16(a1, b1, acc[1][1]);
    acc[1][2] = wmma_f16(a1, b2, acc[1][2]);
    acc[1][3] = wmma_f16(a1, b3, acc[1][3]);
  }

  // C layout: VGPR r, lane -> M = r + 8*sel, N = lane&15
#pragma unroll
  for (int i = 0; i < 2; i++) {
#pragma unroll
    for (int j = 0; j < 4; j++) {
      int col = n0 + j * 16 + lm;
      if (MODE == 0) {
#pragma unroll
        for (int r = 0; r < 8; r++) {
          int m = m0 + i * 16 + r + 8 * sel;
          oh[(size_t)m * CD + col] = (f16)acc[i][j][r];
        }
      } else if (MODE == 1) {
        int h = col >> 6, hd = col & 63;
#pragma unroll
        for (int r = 0; r < 8; r++) {
          int row = m0 + i * 16 + r + 8 * sel;
          int b = row >> 11, s = row & (CS - 1);
          oh[(((size_t)b * CH + h) * CHD + hd) * CS + s] = (f16)acc[i][j][r];
        }
      } else {
        float bv = bias[col];
#pragma unroll
        for (int r = 0; r < 8; r++) {
          int m = m0 + i * 16 + r + 8 * sel;
          of[(size_t)m * CD + col] = acc[i][j][r] + bv;
        }
      }
    }
  }
}

// ---------------------------------------------------------------------------
// In-place RoPE on Qh and Kh (f16). One thread per (tensor, b, s, h, pair j).
// ---------------------------------------------------------------------------
__global__ __launch_bounds__(256) void rope_kernel(f16* __restrict__ Qh,
                                                   f16* __restrict__ Kh) {
  int i = blockIdx.x * blockDim.x + threadIdx.x;   // 2 * B*S*H*32 = 2^22
  int t = i >> 21;
  int r = i & ((1 << 21) - 1);
  int j = r & 31;  r >>= 5;
  int h = r & 15;  r >>= 4;
  int s = r & (CS - 1); r >>= 11;
  int b = r;

  f16* p = (t ? Kh : Qh) + ((size_t)(b * CS + s) * CD + h * CHD + 2 * j);
  // inv_freq = 10000^(-2j/64) = exp(-2j * ln(10000)/64)
  float inv = __expf(-(float)(2 * j) * (9.210340371976184f / 64.0f));
  float fr = (float)s * inv;
  float c = __cosf(fr), sn = __sinf(fr);
  float x1 = (float)p[0], x2 = (float)p[1];
  p[0] = (f16)(x1 * c - x2 * sn);
  p[1] = (f16)(x1 * sn + x2 * c);
}

// ---------------------------------------------------------------------------
// Flash attention with relative bias.
// Block = 8 waves sharing (b,h); each wave owns one 16-row q-tile.
// K/V tiles for each 32-key chunk are staged cooperatively into LDS with
// GLOBAL_LOAD_ASYNC_TO_LDS_B128 (ASYNCcnt, double-buffered). Row-sums of P
// come from an extra WMMA against a ones-matrix (replaces 32 ds_bpermutes).
// ---------------------------------------------------------------------------
__global__ __launch_bounds__(256) void attn_kernel(const f16* __restrict__ Qh,
                                                   const f16* __restrict__ Kh,
                                                   const f16* __restrict__ Vt,
                                                   const int* __restrict__ amask,
                                                   const float* __restrict__ rbias,
                                                   f16* __restrict__ Oh) {
  __shared__ f16 kbuf[2][32 * 64];   // [key 0..31][hd 0..63], 4KB each
  __shared__ f16 vbuf[2][64 * 32];   // [hd 0..63][key 0..31], 4KB each
  __shared__ f16 plds[8 * 16 * 32];  // per-wave 16x32 P tile

  int tid  = threadIdx.x;
  int lane = tid & 31;
  int w    = tid >> 5;
  int wid  = blockIdx.x * 8 + w;         // 0 .. B*H*(S/16)-1 = 4095
  int qt = wid & 127;
  int bh = wid >> 7;
  int h = bh & (CH - 1), b = bh >> 4;
  int q0 = qt * 16;
  int lm = lane & 15, sel = lane >> 4;
  const float scale = 0.125f;  // 1/sqrt(64)

  const f16* kpix  = Kh + (size_t)b * CS * CD + h * CHD;      // [key][hd]
  const f16* vbase = Vt + ((size_t)(b * CH + h) * CHD) * CS;  // [hd][key]
  const float* bbase = rbias + ((size_t)h * CS + q0) * CS;
  const int* mbase = amask + b * CS;
  f16* pw = plds + w * 16 * 32;

  // cooperative async-copy slot for this thread
  int krow = tid >> 3, kseg = tid & 7;   // 32 rows x 8 x 16B
  int vrow = tid >> 2, vseg = tid & 3;   // 64 rows x 4 x 16B

  // Q as A-matrix, two K-dim halves (hd 0..31, 32..63)
  const f16* qrow = Qh + ((size_t)(b * CS + q0 + lm) * CD + h * CHD);
  v16h Qa[2];
#pragma unroll
  for (int t = 0; t < 2; t++) {
    Qa[t] = *(const v16h*)(qrow + t * 32 + sel * 16);
  }

  const v16h ones = {(f16)1,(f16)1,(f16)1,(f16)1,(f16)1,(f16)1,(f16)1,(f16)1,
                     (f16)1,(f16)1,(f16)1,(f16)1,(f16)1,(f16)1,(f16)1,(f16)1};

  float mrow[8], lrow[8];
  v8f O[4];
#pragma unroll
  for (int r = 0; r < 8; r++) { mrow[r] = -INFINITY; lrow[r] = 0.f; }
#pragma unroll
  for (int j = 0; j < 4; j++) O[j] = (v8f){0.f,0.f,0.f,0.f,0.f,0.f,0.f,0.f};

  const int NCHUNK = CS / 32;  // 64

  // prologue: stage chunk 0 into buffer 0
  async_b128(kpix + (size_t)(0 + krow) * CD + kseg * 8, &kbuf[0][krow * 64 + kseg * 8]);
  async_b128(vbase + (size_t)vrow * CS + 0 + vseg * 8, &vbuf[0][vrow * 32 + vseg * 8]);

  for (int ic = 0; ic < NCHUNK; ic++) {
    int kb = ic * 32;
    int cur = ic & 1;

    if (ic + 1 < NCHUNK) {
      int kn = kb + 32;
      async_b128(kpix + (size_t)(kn + krow) * CD + kseg * 8,
                 &kbuf[cur ^ 1][krow * 64 + kseg * 8]);
      async_b128(vbase + (size_t)vrow * CS + kn + vseg * 8,
                 &vbuf[cur ^ 1][vrow * 32 + vseg * 8]);
      // allow the 2 prefetch ops to stay outstanding; chunk `ic` is done
      asm volatile("s_wait_asynccnt 0x2" ::: "memory");
    } else {
      asm volatile("s_wait_asynccnt 0x0" ::: "memory");
    }
    __syncthreads();

    const f16* kl = kbuf[cur];
    const f16* vl = vbuf[cur];

    // ---- scores: two 16-key tiles, K-dim = 64 (2 WMMAs each) ----
    float sc[2][8];
#pragma unroll
    for (int t = 0; t < 2; t++) {
      const f16* krow_l = kl + (16 * t + lm) * 64;
      v16h b0 = *(const v16h*)(krow_l + sel * 16);       // hd 0..31 slice
      v16h b1 = *(const v16h*)(krow_l + 32 + sel * 16);  // hd 32..63 slice
      v8f s_ = {0.f, 0.f, 0.f, 0.f, 0.f, 0.f, 0.f, 0.f};
      s_ = wmma_f16(Qa[0], b0, s_);
      s_ = wmma_f16(Qa[1], b1, s_);
      int key = kb + 16 * t + lm;
      int mv = mbase[key];
      const float* bb = bbase + key;
#pragma unroll
      for (int r = 0; r < 8; r++) {
        int m = r + 8 * sel;
        float v = s_[r] * scale + bb[(size_t)m * CS];
        sc[t][r] = mv ? v : -INFINITY;
      }
    }

    // ---- online softmax: row max via 16-lane xor-shuffle tree ----
    float alpha[8];
#pragma unroll
    for (int r = 0; r < 8; r++) {
      float rm = fmaxf(sc[0][r], sc[1][r]);
#pragma unroll
      for (int off = 1; off < 16; off <<= 1)
        rm = fmaxf(rm, __shfl_xor(rm, off, 32));
      float mn = fmaxf(mrow[r], rm);
      float al = __expf(mrow[r] - mn);
      float p0 = __expf(sc[0][r] - mn);
      float p1 = __expf(sc[1][r] - mn);
      mrow[r] = mn;
      alpha[r] = al;
      // stash P tile (C layout: row = r+8*sel, cols 16t+lm)
      pw[(r + 8 * sel) * 32 + lm]      = (f16)p0;
      pw[(r + 8 * sel) * 32 + 16 + lm] = (f16)p1;
    }
#pragma unroll
    for (int j = 0; j < 4; j++)
#pragma unroll
      for (int r = 0; r < 8; r++) O[j][r] *= alpha[r];

    // C-layout -> A-layout through LDS (same-wave DS RAW; be explicit)
    asm volatile("s_wait_dscnt 0" ::: "memory");
    v16h Pa = *(const v16h*)(pw + lm * 32 + sel * 16);

    // ---- row-sum of P via WMMA against ones: rs[r] lands per-row in C layout
    v8f rs = {0.f, 0.f, 0.f, 0.f, 0.f, 0.f, 0.f, 0.f};
    rs = wmma_f16(Pa, ones, rs);
#pragma unroll
    for (int r = 0; r < 8; r++) lrow[r] = lrow[r] * alpha[r] + rs[r];

    // ---- PV: B from transposed V tile in LDS ----
#pragma unroll
    for (int j = 0; j < 4; j++) {
      v16h Vb = *(const v16h*)(vl + (j * 16 + lm) * 32 + sel * 16);
      O[j] = wmma_f16(Pa, Vb, O[j]);
    }

    __syncthreads();  // everyone done with buf[cur] before it is re-staged
  }

  // epilogue: normalize and store O (f16) in [B,S,D] layout
#pragma unroll
  for (int r = 0; r < 8; r++) {
    float inv = 1.0f / lrow[r];
    int m = r + 8 * sel;
    size_t obase = (size_t)(b * CS + q0 + m) * CD + h * CHD + lm;
#pragma unroll
    for (int j = 0; j < 4; j++) Oh[obase + 16 * j] = (f16)(O[j][r] * inv);
  }
}

// ---------------------------------------------------------------------------
// Host-side orchestration
// ---------------------------------------------------------------------------
extern "C" void kernel_launch(void* const* d_in, const int* in_sizes, int n_in,
                              void* d_out, int out_size, void* d_ws, size_t ws_size,
                              hipStream_t stream) {
  const float* hs    = (const float*)d_in[0];   // [B,S,D]
  const int*   amask = (const int*)d_in[1];     // [B,S]
  const float* Wq    = (const float*)d_in[2];
  const float* Wk    = (const float*)d_in[3];
  const float* Wv    = (const float*)d_in[4];
  const float* Wo    = (const float*)d_in[5];
  const float* bo    = (const float*)d_in[6];
  const float* rbias = (const float*)d_in[7];   // [H,S,S]
  float* out = (float*)d_out;

  const size_t MB = 1024ull * 1024ull;
  char* ws = (char*)d_ws;
  f16* Xh  = (f16*)(ws + 0);        //  8 MB  [4096,1024]
  f16* Wqh = (f16*)(ws + 8 * MB);   //  2 MB
  f16* Wkh = (f16*)(ws + 10 * MB);  //  2 MB
  f16* Wvh = (f16*)(ws + 12 * MB);  //  2 MB
  f16* Woh = (f16*)(ws + 14 * MB);  //  2 MB
  f16* Qh  = (f16*)(ws + 16 * MB);  //  8 MB  [B,S,D]
  f16* Kh  = (f16*)(ws + 24 * MB);  //  8 MB  [B,S,D]
  f16* Vt  = (f16*)(ws + 32 * MB);  //  8 MB  [B,H,HD,S]
  f16* Oh  = (f16*)(ws + 40 * MB);  //  8 MB  [B,S,D]

  // 1. converts
  cvt_f32_f16<<<(CB * CS * CD / 4 + 255) / 256, 256, 0, stream>>>(hs, Xh, CB * CS * CD / 4);
  cvt_f32_f16<<<(CD * CD / 4 + 255) / 256, 256, 0, stream>>>(Wq, Wqh, CD * CD / 4);
  cvt_f32_f16<<<(CD * CD / 4 + 255) / 256, 256, 0, stream>>>(Wk, Wkh, CD * CD / 4);
  cvt_f32_f16<<<(CD * CD / 4 + 255) / 256, 256, 0, stream>>>(Wv, Wvh, CD * CD / 4);
  cvt_f32_f16<<<(CD * CD / 4 + 255) / 256, 256, 0, stream>>>(Wo, Woh, CD * CD / 4);

  // 2. QKV projections: 128 m-groups x 16 n-groups = 2048 waves / 8 per block
  gemm_wmma<0><<<256, 256, 0, stream>>>(Xh, Wqh, Qh, nullptr, nullptr);
  gemm_wmma<0><<<256, 256, 0, stream>>>(Xh, Wkh, Kh, nullptr, nullptr);
  gemm_wmma<1><<<256, 256, 0, stream>>>(Xh, Wvh, Vt, nullptr, nullptr);

  // 3. RoPE on Q and K (2 * B*S*H*32 = 2^22 threads)
  rope_kernel<<<(1 << 22) / 256, 256, 0, stream>>>(Qh, Kh);

  // 4. flash attention: B*H*(S/16) = 4096 waves / 8 per block
  attn_kernel<<<512, 256, 0, stream>>>(Qh, Kh, Vt, amask, rbias, Oh);

  // 5. output projection with bias -> f32 d_out
  gemm_wmma<2><<<256, 256, 0, stream>>>(Oh, Woh, nullptr, out, bo);
}
